// MoELayer_62474594288232
// MI455X (gfx1250) — matrix operand order repaired and checked by reading, not compiled, not run on hardware
//
#include <hip/hip_runtime.h>
#include <hip/hip_bf16.h>

#define BB   4
#define TT   2048
#define DDIM 2048
#define HDIM 8192
#define HRD  4096
#define NRE  3
#define NTOK (BB * TT)          // 8192 tokens
#define CAP  (NTOK + 384)       // 8576: gathered-row capacity (segments padded to 128)

typedef unsigned short u16;
typedef __attribute__((ext_vector_type(16))) u16    u16x16;
typedef __attribute__((ext_vector_type(8)))  u16    u16x8;
typedef __attribute__((ext_vector_type(16))) __bf16 bf16x16;
typedef __attribute__((ext_vector_type(8)))  float  f32x8;

__device__ __forceinline__ u16 f2bf(float f) {
    unsigned u = __builtin_bit_cast(unsigned, f);
    u += 0x7FFFu + ((u >> 16) & 1u);        // round-to-nearest-even
    return (u16)(u >> 16);
}

// Low 32 bits of a flat shared-aperture pointer == wave-relative LDS byte address.
__device__ __forceinline__ unsigned lds_addr(const void* p) {
    return (unsigned)(uintptr_t)p;
}

// Async copy 16B/lane global -> LDS; INST_OFFSET applies to both addresses.
#define ASYNC_LD16(lds, ga, IMMSTR)                                            \
    asm volatile("global_load_async_to_lds_b128 %0, %1, off offset:" IMMSTR   \
                 :: "v"(lds), "v"(ga) : "memory")

__device__ __forceinline__ void issue64(unsigned ldsoff, const u16* g) {
    unsigned long long ga = (unsigned long long)(uintptr_t)g;
    ASYNC_LD16(ldsoff, ga, "0");
    ASYNC_LD16(ldsoff, ga, "16");
    ASYNC_LD16(ldsoff, ga, "32");
    ASYNC_LD16(ldsoff, ga, "48");
}

__device__ __forceinline__ void wait_async0() {
    asm volatile("s_wait_asynccnt 0x0" ::: "memory");
}

// ---------------- init: gather indices to -1, stats/counters to 0 ----------------
__global__ void init_k(int* __restrict__ gidx, float* __restrict__ ctrlF,
                       int* __restrict__ ctrlI) {
    int i = blockIdx.x * 256 + threadIdx.x;
    if (i < CAP) gidx[i] = -1;
    if (i < 8) { ctrlF[i] = 0.0f; ctrlI[i] = 0; }
}

// ---------------- gating: one wave per token ----------------
__global__ void __launch_bounds__(256) gate_k(
    const float* __restrict__ x, const float* __restrict__ noise,
    const float* __restrict__ gw, int* __restrict__ eidx,
    float* __restrict__ escore, float* __restrict__ ctrlF, int* __restrict__ ctrlI) {
    int wave = threadIdx.x >> 5, lane = threadIdx.x & 31;
    int t = blockIdx.x * 8 + wave;
    if (t >= NTOK) return;
    const float* xr = x + (long)t * DDIM;
    float a0 = 0.f, a1 = 0.f, a2 = 0.f;
    for (int d = lane; d < DDIM; d += 32) {
        float xv = xr[d];
        a0 += xv * gw[d];
        a1 += xv * gw[DDIM + d];
        a2 += xv * gw[2 * DDIM + d];
    }
    for (int off = 16; off >= 1; off >>= 1) {
        a0 += __shfl_xor(a0, off, 32);
        a1 += __shfl_xor(a1, off, 32);
        a2 += __shfl_xor(a2, off, 32);
    }
    if (lane == 0) {
        float l[3] = {a0, a1, a2};
        for (int e = 0; e < 3; e++) {
            float nz = noise[(long)t * 3 + e];
            l[e] += -logf(-logf(nz + 1e-9f) + 1e-9f);
        }
        float m = fmaxf(l[0], fmaxf(l[1], l[2]));
        float e0 = expf(l[0] - m), e1 = expf(l[1] - m), e2 = expf(l[2] - m);
        float inv = 1.0f / (e0 + e1 + e2);
        float p[3] = {e0 * inv, e1 * inv, e2 * inv};
        int best = 0; float bv = p[0];
        if (p[1] > bv) { bv = p[1]; best = 1; }
        if (p[2] > bv) { bv = p[2]; best = 2; }
        eidx[t] = best; escore[t] = bv;
        for (int e = 0; e < 3; e++) {
            atomicAdd(&ctrlF[e], p[e]);
            atomicAdd(&ctrlF[3 + e], p[e] * p[e]);
        }
        atomicAdd(&ctrlI[best], 1);
    }
}

// ---------------- finalize: padded segment offsets + loss scalar ----------------
__global__ void finalize_k(const float* __restrict__ ctrlF, int* __restrict__ ctrlI,
                           float* __restrict__ lossOut) {
    if (threadIdx.x == 0) {
        int off = 0;
        for (int e = 0; e < 3; e++) {
            ctrlI[8 + e] = off;
            int c = ctrlI[e];
            off += (c + 127) & ~127;    // pad to GEMM row-tile (128)
        }
        ctrlI[11] = off;
        float inv = 1.0f / (float)NTOK;
        float loss = 0.0f;
        for (int e = 0; e < 3; e++) loss += (ctrlF[e] * inv) * (ctrlF[3 + e] * inv);
        *lossOut = loss * (float)(NRE * NRE);
    }
}

// ---------------- scatter tokens into per-expert segments ----------------
__global__ void scatter_k(const int* __restrict__ eidx, int* __restrict__ gidx,
                          int* __restrict__ ctrlI) {
    int t = blockIdx.x * 256 + threadIdx.x;
    if (t >= NTOK) return;
    int e = eidx[t];
    int pos = atomicAdd(&ctrlI[4 + e], 1);
    gidx[ctrlI[8 + e] + pos] = t;
}

// ---------------- fp32 -> bf16 elementwise ----------------
__global__ void cvt_k(const float* __restrict__ src, u16* __restrict__ dst, int n) {
    int i = (blockIdx.x * 256 + threadIdx.x) * 4;
    if (i + 3 < n) {
        float4 v = *(const float4*)(src + i);
        dst[i]     = f2bf(v.x); dst[i + 1] = f2bf(v.y);
        dst[i + 2] = f2bf(v.z); dst[i + 3] = f2bf(v.w);
    }
}

// ---------------- fp32 [K][N] -> bf16 [N][K] transpose-convert (batched) ----------
__global__ void __launch_bounds__(256) tcvt_k(const float* __restrict__ src,
                                              u16* __restrict__ dst, int K, int N) {
    __shared__ float tile[32][33];
    long base = (long)blockIdx.z * (long)K * (long)N;
    const float* s = src + base;
    u16* d = dst + base;
    int tx = threadIdx.x, ty = threadIdx.y;            // block (32, 8)
    int n0 = blockIdx.x * 32, k0 = blockIdx.y * 32;
    #pragma unroll
    for (int i = 0; i < 4; i++)
        tile[ty + i * 8][tx] = s[(long)(k0 + ty + i * 8) * N + n0 + tx];
    __syncthreads();
    #pragma unroll
    for (int i = 0; i < 4; i++)
        d[(long)(n0 + ty + i * 8) * K + k0 + tx] = f2bf(tile[tx][ty + i * 8]);
}

// ---------------- tiled bf16 WMMA GEMM: 128x128x64, double-buffered async copies --
// EPI: 0 = relu(acc+bias) -> bf16 store   1 = acc+bias -> f32 store
//      2 = out[token] += score[token]*(acc+bias)
// SEG: rows partitioned into expert segments via segOff; GA: gather A rows via gidx
template <int EPI, int SEG, int GA>
__global__ void __launch_bounds__(256) gemm_k(
    const u16* __restrict__ A, const u16* __restrict__ Wt,
    const float* __restrict__ bias, void* __restrict__ OutP,
    int M, int N, int K, int lda, int ldo,
    const int* __restrict__ gidx, const float* __restrict__ score,
    const int* __restrict__ segOff, long wStride, int bStride) {
    int colBase = blockIdx.x * 128;
    int rowBase = blockIdx.y * 128;
    int e = 0;
    if constexpr (SEG) {
        if (rowBase >= segOff[3]) return;
        if (rowBase >= segOff[2]) e = 2;
        else if (rowBase >= segOff[1]) e = 1;
    } else {
        if (rowBase >= M) return;
    }
    const u16* W = Wt + (long)e * wStride;
    const float* bp = bias + (long)e * bStride;

    constexpr int TSZ = 128 * 80;                    // u16 elems per tile buffer
    __shared__ __align__(32) u16 smem[4 * TSZ];      // [buf][A|B] = 80 KB total

    int tid = threadIdx.x;
    int r2 = tid >> 1, h2 = tid & 1;
    int arow = rowBase + r2;
    int srcRow = arow;
    if constexpr (GA) srcRow = gidx[arow];
    bool aValid = (!GA) || (srcRow >= 0);
    const u16* aRow = A + (long)srcRow * lda + h2 * 32;   // only used if aValid
    const u16* bRow = W + (long)(colBase + r2) * K + h2 * 32;

    unsigned ldsA[2], ldsB[2];
    ldsA[0] = lds_addr(smem + 0 * 2 * TSZ + r2 * 80 + h2 * 32);
    ldsA[1] = lds_addr(smem + 1 * 2 * TSZ + r2 * 80 + h2 * 32);
    ldsB[0] = lds_addr(smem + 0 * 2 * TSZ + TSZ + r2 * 80 + h2 * 32);
    ldsB[1] = lds_addr(smem + 1 * 2 * TSZ + TSZ + r2 * 80 + h2 * 32);

    // zero-fill gathered padding rows once (they are never overwritten by async)
    if (GA && !aValid) {
        u16x16 z;
        #pragma unroll
        for (int q = 0; q < 16; q++) z[q] = 0;
        #pragma unroll
        for (int b = 0; b < 2; b++) {
            u16* w = smem + b * 2 * TSZ + r2 * 80 + h2 * 32;
            *(u16x16*)w = z; *(u16x16*)(w + 16) = z;
        }
    }

    int lane = tid & 31;
    int wv = tid >> 5;
    int wm = wv & 3, wn = wv >> 2;
    int mloc = lane & 15;
    int hi16 = lane >> 4;
    int kqa = hi16 * 16, kqb = hi16 * 16;

    f32x8 acc[2][4];
    #pragma unroll
    for (int i = 0; i < 2; i++)
        #pragma unroll
        for (int j = 0; j < 4; j++)
            #pragma unroll
            for (int q = 0; q < 8; q++) acc[i][j][q] = 0.0f;

    const u16* sAr0[2];
    const u16* sBr0[2];
    #pragma unroll
    for (int b = 0; b < 2; b++) {
        sAr0[b] = smem + b * 2 * TSZ + (wm * 32 + mloc) * 80 + kqa;
        sBr0[b] = smem + b * 2 * TSZ + TSZ + (wn * 64 + mloc) * 80 + kqb;
    }

    auto compute = [&](int buf) {
        #pragma unroll
        for (int kk = 0; kk < 64; kk += 32) {
            bf16x16 aF[2], bF[4];
            #pragma unroll
            for (int i = 0; i < 2; i++) {
                const u16* p = sAr0[buf] + i * 16 * 80 + kk;
                u16x8 lo = *(const u16x8*)p;
                u16x8 hi = *(const u16x8*)(p + 8);
                u16x16 t;
                #pragma unroll
                for (int q = 0; q < 8; q++) { t[q] = lo[q]; t[q + 8] = hi[q]; }
                aF[i] = __builtin_bit_cast(bf16x16, t);
            }
            #pragma unroll
            for (int j = 0; j < 4; j++) {
                const u16* p = sBr0[buf] + j * 16 * 80 + kk;
                bF[j] = __builtin_bit_cast(bf16x16, *(const u16x16*)p);
            }
            #pragma unroll
            for (int i = 0; i < 2; i++)
                #pragma unroll
                for (int j = 0; j < 4; j++)
                    acc[i][j] = __builtin_amdgcn_wmma_f32_16x16x32_bf16(
                        false, aF[i], false, bF[j], (short)0, acc[i][j], false, false);
        }
    };
    auto step = [&](int buf, int kb) {
        wait_async0();          // my async copies into `buf` are in LDS
        __syncthreads();        // everyone's copies done; prev compute on buf^1 done
        int nkb = kb + 64;
        if (nkb < K) {          // prefetch next tile into the other buffer
            int bn = buf ^ 1;
            if (aValid) issue64(ldsA[bn], aRow + nkb);
            issue64(ldsB[bn], bRow + nkb);
        }
        compute(buf);
    };

    // prologue: fill buffer 0 with the first K-tile
    if (aValid) issue64(ldsA[0], aRow);
    issue64(ldsB[0], bRow);
    for (int kb = 0; kb < K; kb += 128) {   // K is a multiple of 128 in all passes
        step(0, kb);
        step(1, kb + 64);
    }

    int mb = hi16 * 8;
    #pragma unroll
    for (int i = 0; i < 2; i++) {
        int gr0 = rowBase + wm * 32 + i * 16 + mb;
        int tok[8]; float sc[8];
        if constexpr (EPI == 2) {
            #pragma unroll
            for (int p = 0; p < 8; p++) {
                tok[p] = gidx[gr0 + p];
                sc[p] = (tok[p] >= 0) ? score[tok[p]] : 0.0f;
            }
        }
        #pragma unroll
        for (int j = 0; j < 4; j++) {
            int gc = colBase + wn * 64 + j * 16 + mloc;
            float bv = bp[gc];
            #pragma unroll
            for (int p = 0; p < 8; p++) {
                float v = acc[i][j][p] + bv;
                if constexpr (EPI == 0) {
                    v = fmaxf(v, 0.0f);
                    ((u16*)OutP)[(long)(gr0 + p) * ldo + gc] = f2bf(v);
                } else if constexpr (EPI == 1) {
                    ((float*)OutP)[(long)(gr0 + p) * ldo + gc] = v;
                } else {
                    if (tok[p] >= 0) {
                        float* o = (float*)OutP + (long)tok[p] * ldo + gc;
                        *o += sc[p] * v;   // segments disjoint per token: no atomics
                    }
                }
            }
        }
    }
}

extern "C" void kernel_launch(void* const* d_in, const int* in_sizes, int n_in,
                              void* d_out, int out_size, void* d_ws, size_t ws_size,
                              hipStream_t stream) {
    const float* x      = (const float*)d_in[0];
    const float* noise  = (const float*)d_in[1];
    const float* gate_w = (const float*)d_in[2];
    const float* w1_s   = (const float*)d_in[3];
    const float* b1_s   = (const float*)d_in[4];
    const float* w2_s   = (const float*)d_in[5];
    const float* b2_s   = (const float*)d_in[6];
    const float* w1_r   = (const float*)d_in[7];
    const float* b1_r   = (const float*)d_in[8];
    const float* w2_r   = (const float*)d_in[9];
    const float* b2_r   = (const float*)d_in[10];
    float* out = (float*)d_out;
    (void)in_sizes; (void)n_in; (void)out_size; (void)ws_size;

    char* ws = (char*)d_ws;
    size_t o = 0;
    auto take = [&](size_t bytes) -> char* {
        char* p = ws + o;
        o += (bytes + 255) & ~(size_t)255;
        return p;
    };
    float* ctrlF = (float*)take(64);
    int*   ctrlI = (int*)take(64);
    int*   eidx  = (int*)take((size_t)NTOK * 4);
    float* escore= (float*)take((size_t)NTOK * 4);
    int*   gidx  = (int*)take((size_t)CAP * 4);
    u16*   xb    = (u16*)take((size_t)NTOK * DDIM * 2);
    u16*   w1sT  = (u16*)take((size_t)DDIM * HDIM * 2);
    u16*   w2sT  = (u16*)take((size_t)HDIM * DDIM * 2);
    u16*   w1rT  = (u16*)take((size_t)NRE * DDIM * HRD * 2);
    u16*   w2rT  = (u16*)take((size_t)NRE * HRD * DDIM * 2);
    u16*   hbuf  = (u16*)take((size_t)NTOK * HDIM * 2);
    u16*   hg    = (u16*)take((size_t)CAP * HRD * 2);

    // gating pipeline
    init_k<<<(CAP + 255) / 256, 256, 0, stream>>>(gidx, ctrlF, ctrlI);
    gate_k<<<NTOK / 8, 256, 0, stream>>>(x, noise, gate_w, eidx, escore, ctrlF, ctrlI);
    finalize_k<<<1, 32, 0, stream>>>(ctrlF, ctrlI, out + (size_t)NTOK * DDIM);
    scatter_k<<<NTOK / 256, 256, 0, stream>>>(eidx, gidx, ctrlI);

    // conversions (weights transposed -> [N][K] bf16; all bf16 weights fit in L2)
    cvt_k<<<(NTOK * DDIM / 4 + 255) / 256, 256, 0, stream>>>(x, xb, NTOK * DDIM);
    dim3 tb(32, 8);
    tcvt_k<<<dim3(HDIM / 32, DDIM / 32, 1),  tb, 0, stream>>>(w1_s, w1sT, DDIM, HDIM);
    tcvt_k<<<dim3(DDIM / 32, HDIM / 32, 1),  tb, 0, stream>>>(w2_s, w2sT, HDIM, DDIM);
    tcvt_k<<<dim3(HRD / 32, DDIM / 32, NRE), tb, 0, stream>>>(w1_r, w1rT, DDIM, HRD);
    tcvt_k<<<dim3(DDIM / 32, HRD / 32, NRE), tb, 0, stream>>>(w2_r, w2rT, HRD, DDIM);

    // shared expert: out = relu(x@W1+b1)@W2 + b2
    gemm_k<0, 0, 0><<<dim3(HDIM / 128, NTOK / 128), 256, 0, stream>>>(
        xb, w1sT, b1_s, hbuf, NTOK, HDIM, DDIM, DDIM, HDIM,
        nullptr, nullptr, nullptr, 0L, 0);
    gemm_k<1, 0, 0><<<dim3(DDIM / 128, NTOK / 128), 256, 0, stream>>>(
        hbuf, w2sT, b2_s, out, NTOK, DDIM, HDIM, HDIM, DDIM,
        nullptr, nullptr, nullptr, 0L, 0);

    // routed experts: gathered-token GEMMs (top-1 => 1/3 of the dense FLOPs)
    gemm_k<0, 1, 1><<<dim3(HRD / 128, CAP / 128), 256, 0, stream>>>(
        xb, w1rT, b1_r, hg, CAP, HRD, DDIM, DDIM, HRD,
        gidx, nullptr, ctrlI + 8, (long)DDIM * HRD, HRD);
    gemm_k<2, 1, 0><<<dim3(DDIM / 128, CAP / 128), 256, 0, stream>>>(
        hg, w2rT, b2_r, out, CAP, DDIM, HRD, HRD, DDIM,
        gidx, escore, ctrlI + 8, (long)HRD * DDIM, DDIM);
}
